// Block_83588653514816
// MI455X (gfx1250) — compile-verified, matrix operand order checked by reading, not run on hardware
//
#include <hip/hip_runtime.h>
#include <hip/hip_bf16.h>

// ---------------------------------------------------------------- constants
constexpr int kB   = 4;
constexpr int kT   = 2048;
constexpr int kD   = 2048;
constexpr int kH   = 16;
constexpr int kDH  = 128;
constexpr int kTD  = kH * kDH;      // 2048
constexpr int kDFF = 8192;
constexpr int kM   = kB * kT;       // 8192 rows for all GEMMs

typedef __bf16 bf16;
typedef bf16  bf16x8  __attribute__((ext_vector_type(8)));
typedef bf16  bf16x16 __attribute__((ext_vector_type(16)));
typedef float f32x8   __attribute__((ext_vector_type(8)));
typedef unsigned int u32x4 __attribute__((ext_vector_type(4)));
typedef int          i32x8 __attribute__((ext_vector_type(8)));
typedef int          i32x4 __attribute__((ext_vector_type(4)));

// ---------------------------------------------------------------- helpers
__device__ inline f32x8 zero8() {
  f32x8 z;
#pragma unroll
  for (int i = 0; i < 8; ++i) z[i] = 0.0f;
  return z;
}

// 16-bit WMMA fragment loader from global (A 16x32 or B 32x16, contraction
// dim innermost). Per ISA 7.12.2: lane L -> row L%16; 8 elems at
// k0+((L&16)?8:0), 8 more at +16.
__device__ inline bf16x16 load_frag(const bf16* __restrict__ p, int ld,
                                    int row0, int k0) {
  int lane = threadIdx.x & 31;
  const bf16* base =
      p + (long)(row0 + (lane & 15)) * ld + k0 + ((lane & 16) ? 8 : 0);
  bf16x8 lo = *(const bf16x8*)(base);
  bf16x8 hi = *(const bf16x8*)(base + 16);
  bf16x16 r;
#pragma unroll
  for (int i = 0; i < 8; ++i) { r[i] = lo[i]; r[8 + i] = hi[i]; }
  return r;
}

// Same fragment pattern but from an LDS tile with padded row pitch.
constexpr int LDSROW = 40;  // elements: 32 data + 4-DWORD TDM pad -> 80 bytes
__device__ inline bf16x16 load_frag_lds(const bf16* p, int row0) {
  int lane = threadIdx.x & 31;
  const bf16* base = p + (row0 + (lane & 15)) * LDSROW + ((lane & 16) ? 8 : 0);
  bf16x8 lo = *(const bf16x8*)(base);
  bf16x8 hi = *(const bf16x8*)(base + 16);
  bf16x16 r;
#pragma unroll
  for (int i = 0; i < 8; ++i) { r[i] = lo[i]; r[8 + i] = hi[i]; }
  return r;
}

__device__ inline f32x8 wmma_bf16(bf16x16 a, bf16x16 b, f32x8 c) {
  return __builtin_amdgcn_wmma_f32_16x16x32_bf16(false, a, false, b,
                                                 (short)0, c, false, false);
}

__device__ inline float hmax16(float x) {   // reduce over a 16-lane half
#pragma unroll
  for (int off = 8; off; off >>= 1) x = fmaxf(x, __shfl_xor(x, off, 32));
  return x;
}
__device__ inline float hsum16(float x) {
#pragma unroll
  for (int off = 8; off; off >>= 1) x += __shfl_xor(x, off, 32);
  return x;
}

// ------------------------------------------------------------ TDM (ISA ch8)
// 2-D tile load: tile_dim0 = 32 elems (bf16), tile_dim1 = rows, row pitch =
// row_stride_elems. LDS destination padded +4 DWORDs every 16 DWORDs so the
// staged tile has an 80-byte row pitch (bank-conflict-free b128 frag reads).
// Toolchain here is the clang-23 lane: 6-arg builtin
//   (u32x4 g0, i32x8 g1, i32x4 g2, i32x4 g3, i32x8 extra, i32 cpol)
__device__ inline void tdm_load_2d(unsigned lds_off, const void* gptr,
                                   int rows, long row_stride_elems) {
  unsigned long long ga = (unsigned long long)(size_t)gptr;
  u32x4 g0;
  g0[0] = 1u;                                    // count=1 (user descriptor)
  g0[1] = lds_off;                               // lds_addr (bytes)
  g0[2] = (unsigned)ga;                          // global_addr[31:0]
  g0[3] = (unsigned)((ga >> 32) & 0x01ffffffull) // global_addr[56:32]
          | (2u << 30);                          // type=2 ("image")
  i32x8 g1;
  g1[0] = (int)((1u << 16)      // data_size = 2 bytes
                | (1u << 20)    // pad_enable
                | (3u << 22)    // pad_interval: every 16 DWORDs (64B)
                | (3u << 25));  // pad_amount: 4 DWORDs (16B)
  unsigned td0 = 32u;                 // tensor_dim0 (elements)
  unsigned td1 = (unsigned)rows;      // tensor_dim1
  g1[1] = (int)((td0 & 0xffffu) << 16);                    // [15:0] bar addr=0
  g1[2] = (int)(((td0 >> 16) & 0xffffu) | ((td1 & 0xffffu) << 16));
  g1[3] = (int)(((td1 >> 16) & 0xffffu) | (32u << 16));    // tile_dim0 = 32
  g1[4] = (int)((unsigned)rows & 0xffffu);                 // tile_dim1; dim2=0
  unsigned long long s0 = (unsigned long long)row_stride_elems;
  g1[5] = (int)(unsigned)(s0 & 0xffffffffull);             // dim0_stride lo
  g1[6] = (int)(unsigned)((s0 >> 32) & 0xffffull);         // hi; dim1_stride=0
  g1[7] = 0;
  i32x4 z4;
  z4[0] = z4[1] = z4[2] = z4[3] = 0;             // groups 2/3 unused (2-D)
  i32x8 z8;
#pragma unroll
  for (int i = 0; i < 8; ++i) z8[i] = 0;
  __builtin_amdgcn_tensor_load_to_lds(g0, g1, z4, z4, z8, 0);
}

// ---------------------------------------------------------------- fp32->bf16
__global__ void cvt_kernel(const float* __restrict__ in, bf16* __restrict__ out,
                           long n) {
  long i = (long)blockIdx.x * blockDim.x + threadIdx.x;
  long stride = (long)gridDim.x * blockDim.x;
  for (; i < n; i += stride) out[i] = (bf16)in[i];
}

// ---------------------------------------------------------------- RMSNorm
__global__ __launch_bounds__(256) void rmsnorm_kernel(
    const float* __restrict__ x, const float* __restrict__ g,
    bf16* __restrict__ out) {
  __shared__ float red[8];
  long row = blockIdx.x;
  const float* xr = x + row * (long)kD;
  float ss = 0.0f;
#pragma unroll
  for (int j = 0; j < kD / 256; ++j) {
    float v = xr[threadIdx.x + j * 256];
    ss += v * v;
  }
#pragma unroll
  for (int off = 16; off; off >>= 1) ss += __shfl_xor(ss, off, 32);
  if ((threadIdx.x & 31) == 0) red[threadIdx.x >> 5] = ss;
  __syncthreads();
  float tot = 0.0f;
#pragma unroll
  for (int i = 0; i < 8; ++i) tot += red[i];
  float rms = rsqrtf(tot / (float)kD + 1e-6f);
  bf16* orow = out + row * (long)kD;
#pragma unroll
  for (int j = 0; j < kD / 256; ++j) {
    int i = threadIdx.x + j * 256;
    orow[i] = (bf16)(xr[i] * rms * g[i]);
  }
}

// ---------------------------------------------------------------- WMMA GEMM
// C[M,N] = A[M,K] @ W[N,K]^T.  Block: 256 thr = 8 waves (2 M x 4 N), wave
// tile 32x64, block tile 64x256.  Per K-step of 32 the TDM stages the whole
// 256x32 W tile + 64x32 A tile into LDS (double buffered, padded rows);
// waves read fragments with ds_load_b128.  Epilogues:
//   0: bf16 store   1: QKV scatter   2: fp32 residual add   3: silu(gate)*up
template <int EPI>
__global__ __launch_bounds__(256) void gemm_kernel(
    const bf16* __restrict__ A, const bf16* __restrict__ W, int M, int N,
    int K, const float* __restrict__ res, float* __restrict__ outf,
    bf16* __restrict__ outb, const bf16* __restrict__ gate,
    bf16* __restrict__ qb, bf16* __restrict__ kb, bf16* __restrict__ vtb) {
  __shared__ bf16 smW[2][256 * LDSROW];   // 2 x 20 KB
  __shared__ bf16 smA[2][64 * LDSROW];    // 2 x  5 KB
  int w = threadIdx.x >> 5;
  int lane = threadIdx.x & 31;
  int mw = w >> 2, nw = w & 3;
  int rblk = blockIdx.y * 64;
  int cblk = blockIdx.x * 256;
  int r0 = mw * 32;   // wave row offset within block tile
  int c0 = nw * 64;   // wave col offset within block tile

  f32x8 acc[2][4];
#pragma unroll
  for (int i = 0; i < 2; ++i)
#pragma unroll
    for (int j = 0; j < 4; ++j) acc[i][j] = zero8();

  const bf16* Ablk = A + (long)rblk * K;
  const bf16* Wblk = W + (long)cblk * K;
  int nk = K / 32;

  if (w == 0) {
    tdm_load_2d((unsigned)(size_t)&smW[0][0], Wblk, 256, K);
    tdm_load_2d((unsigned)(size_t)&smA[0][0], Ablk, 64, K);
  }

  for (int i = 0; i < nk; ++i) {
    if (w == 0) __builtin_amdgcn_s_wait_tensorcnt(0);
    __syncthreads();                     // tile i visible; tile i-1 consumed
    int buf = i & 1;
    if (w == 0 && i + 1 < nk) {          // overlap next DMA with compute
      tdm_load_2d((unsigned)(size_t)&smW[buf ^ 1][0], Wblk + (i + 1) * 32,
                  256, K);
      tdm_load_2d((unsigned)(size_t)&smA[buf ^ 1][0], Ablk + (i + 1) * 32,
                  64, K);
    }
    bf16x16 a0 = load_frag_lds(&smA[buf][0], r0);
    bf16x16 a1 = load_frag_lds(&smA[buf][0], r0 + 16);
#pragma unroll
    for (int j = 0; j < 4; ++j) {
      bf16x16 bfr = load_frag_lds(&smW[buf][0], c0 + 16 * j);
      acc[0][j] = wmma_bf16(a0, bfr, acc[0][j]);
      acc[1][j] = wmma_bf16(a1, bfr, acc[1][j]);
    }
  }

  int lh = (lane & 16) ? 8 : 0;
  int lc = lane & 15;
#pragma unroll
  for (int ms = 0; ms < 2; ++ms)
#pragma unroll
    for (int j = 0; j < 4; ++j)
#pragma unroll
      for (int v = 0; v < 8; ++v) {
        int m = rblk + r0 + ms * 16 + v + lh;
        int n = cblk + c0 + j * 16 + lc;
        float val = acc[ms][j][v];
        long idx = (long)m * N + n;
        if (EPI == 0) {
          outb[idx] = (bf16)val;
        } else if (EPI == 2) {
          outf[idx] = val + res[idx];
        } else if (EPI == 3) {
          float g = (float)gate[idx];
          float s = g / (1.0f + __expf(-g));
          outb[idx] = (bf16)(s * val);
        } else {  // EPI == 1: QKV scatter
          int bb = m >> 11;             // row = b*T + t, T=2048
          int t = m & (kT - 1);
          int which = n >> 11;          // 0:q 1:k 2:v  (TD = 2048)
          int col = n & (kTD - 1);
          int hh = col >> 7;            // DH = 128
          int dh = col & (kDH - 1);
          if (which == 0)
            qb[(((long)(bb * kH + hh)) * kT + t) * kDH + dh] = (bf16)val;
          else if (which == 1)
            kb[(((long)(bb * kH + hh)) * kT + t) * kDH + dh] = (bf16)val;
          else
            vtb[((long)(bb * kH + hh)) * kDH * kT + (long)dh * kT + t] =
                (bf16)val;
        }
      }
}

// ---------------------------------------------------------------- attention
// One wave per (b, h, 16-query tile): flash attention over 32-key blocks.
constexpr int PSTR = 36;  // padded LDS row stride (floats)

__global__ __launch_bounds__(256) void attn_kernel(
    const bf16* __restrict__ q, const bf16* __restrict__ k,
    const bf16* __restrict__ vt, bf16* __restrict__ o) {
  __shared__ float ldsP[8][16 * PSTR];
  int w = threadIdx.x >> 5;
  int lane = threadIdx.x & 31;
  int gid = blockIdx.x * 8 + w;
  int qt = gid & (kT / 16 - 1);   // 128 query tiles
  int bh = gid >> 7;
  int b = bh >> 4;                // H = 16
  int h = bh & 15;
  const bf16* qh = q + (long)bh * kT * kDH;
  const bf16* kh = k + (long)bh * kT * kDH;
  const bf16* vth = vt + (long)bh * kDH * kT;
  int q0 = qt * 16;

  bf16x16 qf[4];
#pragma unroll
  for (int kc = 0; kc < 4; ++kc) qf[kc] = load_frag(qh, kDH, q0, kc * 32);

  f32x8 oacc[8];
#pragma unroll
  for (int nt = 0; nt < 8; ++nt) oacc[nt] = zero8();
  float mrow[8], lrow[8], alpha[8];
#pragma unroll
  for (int v = 0; v < 8; ++v) { mrow[v] = -3.0e38f; lrow[v] = 0.0f; }

  int lh = (lane & 16) ? 8 : 0;
  int lc = lane & 15;
  float* myP = ldsP[w];
  const float scale = 0.08838834764831845f;  // 1/sqrt(128)
  int kend = ((q0 >> 5) + 1) * 32;           // causal key-block limit

  for (int jb = 0; jb < kend; jb += 32) {
    f32x8 s0 = zero8(), s1 = zero8();
#pragma unroll
    for (int kc = 0; kc < 4; ++kc) {
      bf16x16 k0f = load_frag(kh, kDH, jb, kc * 32);
      bf16x16 k1f = load_frag(kh, kDH, jb + 16, kc * 32);
      s0 = wmma_bf16(qf[kc], k0f, s0);
      s1 = wmma_bf16(qf[kc], k1f, s1);
    }
#pragma unroll
    for (int v = 0; v < 8; ++v) {
      int m = v + lh;
      int query = q0 + m;
      float a = s0[v] * scale;
      if (jb + lc > query) a = -3.0e38f;
      float bb = s1[v] * scale;
      if (jb + 16 + lc > query) bb = -3.0e38f;
      float mx = hmax16(fmaxf(a, bb));
      float mnew = fmaxf(mrow[v], mx);
      float al = __expf(mrow[v] - mnew);
      mrow[v] = mnew;
      float p0 = __expf(a - mnew);
      float p1 = __expf(bb - mnew);
      lrow[v] = lrow[v] * al + hsum16(p0 + p1);
      alpha[v] = al;
      myP[m * PSTR + lc] = p0;          // C-layout -> LDS (16x32 tile)
      myP[m * PSTR + 16 + lc] = p1;
    }
    asm volatile("" ::: "memory");      // same-wave LDS ops are in-order
    bf16x16 pf;                         // LDS -> A-layout fragment
#pragma unroll
    for (int i = 0; i < 8; ++i) {
      pf[i]     = (bf16)myP[lc * PSTR + lh + i];
      pf[8 + i] = (bf16)myP[lc * PSTR + 16 + lh + i];
    }
#pragma unroll
    for (int nt = 0; nt < 8; ++nt) {
#pragma unroll
      for (int v = 0; v < 8; ++v) oacc[nt][v] *= alpha[v];
      bf16x16 vf = load_frag(vth, kT, nt * 16, jb);
      oacc[nt] = wmma_bf16(pf, vf, oacc[nt]);
    }
  }
#pragma unroll
  for (int v = 0; v < 8; ++v) lrow[v] = 1.0f / lrow[v];
#pragma unroll
  for (int nt = 0; nt < 8; ++nt)
#pragma unroll
    for (int v = 0; v < 8; ++v) {
      int m = v + lh;
      long idx = ((long)b * kT + q0 + m) * kTD + (long)h * kDH + nt * 16 + lc;
      o[idx] = (bf16)(oacc[nt][v] * lrow[v]);
    }
}

// ---------------------------------------------------------------- workspace
constexpr long NE_WQKV = 3L * kTD * kD;      // 12,582,912
constexpr long NE_WOUT = (long)kD * kTD;     //  4,194,304
constexpr long NE_WFF  = (long)kDFF * kD;    // 16,777,216
constexpr long NE_H    = (long)kM * kD;      // 16,777,216
constexpr long NE_HD   = (long)kM * kTD;     // 16,777,216 (q/k/vt/o each)
constexpr long NE_FF   = (long)kM * kDFF;    // 67,108,864

constexpr size_t OFF_WQKV = 0;
constexpr size_t OFF_WOUT = OFF_WQKV + NE_WQKV * 2;
constexpr size_t OFF_WGATE = OFF_WOUT + NE_WOUT * 2;
constexpr size_t OFF_WUP = OFF_WGATE + NE_WFF * 2;
constexpr size_t OFF_WDOWN = OFF_WUP + NE_WFF * 2;
constexpr size_t OFF_H = OFF_WDOWN + NE_WFF * 2;
constexpr size_t OFF_XMID = OFF_H + NE_H * 2;
constexpr size_t OFF_SCR = OFF_XMID + NE_H * 4;
// attention phase (aliases MLP gate region):
constexpr size_t OFF_Q = OFF_SCR;
constexpr size_t OFF_K = OFF_Q + NE_HD * 2;
constexpr size_t OFF_VT = OFF_K + NE_HD * 2;
constexpr size_t OFF_O = OFF_VT + NE_HD * 2;
// MLP phase:
constexpr size_t OFF_GATE = OFF_SCR;
constexpr size_t OFF_ACT = OFF_GATE + NE_FF * 2;

// ---------------------------------------------------------------- launch
extern "C" void kernel_launch(void* const* d_in, const int* in_sizes, int n_in,
                              void* d_out, int out_size, void* d_ws,
                              size_t ws_size, hipStream_t stream) {
  const float* x      = (const float*)d_in[0];
  const float* w_qkv  = (const float*)d_in[1];
  const float* w_out  = (const float*)d_in[2];
  const float* w_gate = (const float*)d_in[3];
  const float* w_up   = (const float*)d_in[4];
  const float* w_down = (const float*)d_in[5];
  const float* ln1    = (const float*)d_in[6];
  const float* ln2    = (const float*)d_in[7];
  float* out = (float*)d_out;
  char* ws = (char*)d_ws;

  bf16* wqkv_b  = (bf16*)(ws + OFF_WQKV);
  bf16* wout_b  = (bf16*)(ws + OFF_WOUT);
  bf16* wgate_b = (bf16*)(ws + OFF_WGATE);
  bf16* wup_b   = (bf16*)(ws + OFF_WUP);
  bf16* wdown_b = (bf16*)(ws + OFF_WDOWN);
  bf16* h_b     = (bf16*)(ws + OFF_H);
  float* xmid   = (float*)(ws + OFF_XMID);
  bf16* qb      = (bf16*)(ws + OFF_Q);
  bf16* kb      = (bf16*)(ws + OFF_K);
  bf16* vtb     = (bf16*)(ws + OFF_VT);
  bf16* ob      = (bf16*)(ws + OFF_O);
  bf16* gate_b  = (bf16*)(ws + OFF_GATE);
  bf16* act_b   = (bf16*)(ws + OFF_ACT);

  // 1. weights -> bf16
  cvt_kernel<<<4096, 256, 0, stream>>>(w_qkv, wqkv_b, NE_WQKV);
  cvt_kernel<<<4096, 256, 0, stream>>>(w_out, wout_b, NE_WOUT);
  cvt_kernel<<<4096, 256, 0, stream>>>(w_gate, wgate_b, NE_WFF);
  cvt_kernel<<<4096, 256, 0, stream>>>(w_up, wup_b, NE_WFF);
  cvt_kernel<<<4096, 256, 0, stream>>>(w_down, wdown_b, NE_WFF);

  // 2. h = rmsnorm(x, ln1)
  rmsnorm_kernel<<<kM, 256, 0, stream>>>(x, ln1, h_b);

  // 3. qkv = h @ w_qkv^T, scattered into q/k (B,H,T,DH) and v^T (B,H,DH,T)
  {
    dim3 g(3 * kTD / 256, kM / 64);
    gemm_kernel<1><<<g, 256, 0, stream>>>(h_b, wqkv_b, kM, 3 * kTD, kD,
                                          nullptr, nullptr, nullptr, nullptr,
                                          qb, kb, vtb);
  }

  // 4. flash attention -> o (bf16, [B,T,TD])
  attn_kernel<<<(kB * kH * (kT / 16)) / 8, 256, 0, stream>>>(qb, kb, vtb, ob);

  // 5. xmid = x + o @ w_out^T
  {
    dim3 g(kD / 256, kM / 64);
    gemm_kernel<2><<<g, 256, 0, stream>>>(ob, wout_b, kM, kD, kTD, x, xmid,
                                          nullptr, nullptr, nullptr, nullptr,
                                          nullptr);
  }

  // 6. h = rmsnorm(xmid, ln2)
  rmsnorm_kernel<<<kM, 256, 0, stream>>>(xmid, ln2, h_b);

  // 7. gate = h @ w_gate^T (bf16)
  {
    dim3 g(kDFF / 256, kM / 64);
    gemm_kernel<0><<<g, 256, 0, stream>>>(h_b, wgate_b, kM, kDFF, kD, nullptr,
                                          nullptr, gate_b, nullptr, nullptr,
                                          nullptr, nullptr);
    // 8. act = silu(gate) * (h @ w_up^T)
    gemm_kernel<3><<<g, 256, 0, stream>>>(h_b, wup_b, kM, kDFF, kD, nullptr,
                                          nullptr, act_b, gate_b, nullptr,
                                          nullptr, nullptr);
  }

  // 9. out = xmid + act @ w_down^T
  {
    dim3 g(kD / 256, kM / 64);
    gemm_kernel<2><<<g, 256, 0, stream>>>(act_b, wdown_b, kM, kD, kDFF, xmid,
                                          out, nullptr, nullptr, nullptr,
                                          nullptr, nullptr);
  }
}
